// AttentionHeads_63101659513117
// MI455X (gfx1250) — matrix operand order repaired and checked by reading, not compiled, not run on hardware
//
#include <hip/hip_runtime.h>
#include <hip/hip_bf16.h>

typedef __bf16 bf16_t;
typedef __attribute__((ext_vector_type(16))) __bf16 v16bf;
typedef __attribute__((ext_vector_type(8)))  __bf16 v8bf;
typedef __attribute__((ext_vector_type(8)))  float  v8f;

// ---- Tensor Data Mover availability / arity detection -----------------------
#if defined(__has_builtin)
# if __has_builtin(__builtin_amdgcn_tensor_load_to_lds)
#  define HAVE_TDM 1
# endif
#endif
#ifndef HAVE_TDM
# define HAVE_TDM 0
#endif
#if __has_include(<hip/amd_detail/amd_gfx1250_TDM.h>)
# define TDM_6ARG 1
#else
# define TDM_6ARG 0
#endif

__device__ __forceinline__ v16bf join8(v8bf lo, v8bf hi) {
  v16bf r;
#pragma unroll
  for (int i = 0; i < 8; ++i) { r[i] = lo[i]; r[8 + i] = hi[i]; }
  return r;
}

#if HAVE_TDM
typedef __attribute__((ext_vector_type(4))) unsigned int v4u;
typedef __attribute__((ext_vector_type(8))) int v8i;
typedef __attribute__((ext_vector_type(4))) int v4i;

// Issue one 2-D TDM tile load: global (row-major, stride0 elems/row) -> LDS.
// All arguments must be wave-uniform. data_size = 2 bytes (bf16).
__device__ __forceinline__ void tdm_load_2d(const void* gptr, unsigned lds_off,
                                            unsigned dim0, unsigned dim1,
                                            unsigned tile0, unsigned tile1,
                                            unsigned long long stride0) {
  unsigned long long ga = (unsigned long long)(uintptr_t)gptr;
  v4u g0;
  g0[0] = 1u;                                   // count=1, user mode
  g0[1] = lds_off;                              // lds_addr (bytes)
  g0[2] = (unsigned)ga;                         // global_addr[31:0]
  g0[3] = ((unsigned)(ga >> 32) & 0x01FFFFFFu)  // global_addr[56:32]
          | 0x80000000u;                        // type = 2 ("image")
  v8i g1;
  g1[0] = (int)(1u << 16);                      // data_size = 1 -> 2 bytes
  g1[1] = (int)((dim0 & 0xFFFFu) << 16);        // tensor_dim0[15:0]
  g1[2] = (int)(((dim0 >> 16) & 0xFFFFu) | ((dim1 & 0xFFFFu) << 16));
  g1[3] = (int)(((dim1 >> 16) & 0xFFFFu) | ((tile0 & 0xFFFFu) << 16));
  g1[4] = (int)(tile1 & 0xFFFFu);               // tile_dim1 (tile_dim2 = 0)
  g1[5] = (int)(unsigned)(stride0 & 0xFFFFFFFFull);        // stride0[31:0]
  g1[6] = (int)(unsigned)((stride0 >> 32) & 0xFFFFull);    // stride0[47:32]
  g1[7] = 0;
  v4i z = {0, 0, 0, 0};
#if TDM_6ARG
  v8i z8 = {0, 0, 0, 0, 0, 0, 0, 0};
  __builtin_amdgcn_tensor_load_to_lds(g0, g1, z, z, z8, 0);
#else
  __builtin_amdgcn_tensor_load_to_lds(g0, g1, z, z, 0);
#endif
}
#endif  // HAVE_TDM

// ---------------------------------------------------------------- conversion
__global__ void cvt_f32_bf16(const float* __restrict__ src,
                             bf16_t* __restrict__ dst, int n) {
  int i = blockIdx.x * blockDim.x + threadIdx.x;
  if (i < n) dst[i] = (bf16_t)src[i];
}

// ---------------------------------------------------------------- projection
// y = x @ W^T.  X: [8192,1024] bf16.  W: [1024,1024] bf16 (row = out feature).
// Register-blocked: each wave owns a 32x64 output tile (2 A-frags x 4 B-frags,
// 8 WMMAs per 32-deep K step -> 0.75 fragment loads per WMMA).
// Output head-split bf16: mode 0 -> [B][H][T][64]; mode 1 -> [B][H][64][T].
__global__ void __launch_bounds__(128)
proj_gemm(const bf16_t* __restrict__ X, const bf16_t* __restrict__ W,
          bf16_t* __restrict__ Y, int mode) {
  const int lane = threadIdx.x & 31;
  const int wv   = threadIdx.x >> 5;
  const int h    = lane >> 4;          // lane half (ISA 16-bit frag layout)
  const int l16  = lane & 15;
  const int K    = 1024;
  const int mbase = blockIdx.x * 32;
  const int nbase = blockIdx.y * 256 + wv * 64;

  const bf16_t* xr[2];
#pragma unroll
  for (int mi = 0; mi < 2; ++mi)
    xr[mi] = X + (size_t)(mbase + mi * 16 + l16) * K;
  const bf16_t* wr[4];
#pragma unroll
  for (int nt = 0; nt < 4; ++nt)
    wr[nt] = W + (size_t)(nbase + nt * 16 + l16) * K;

  v8f acc[2][4] = {};
  for (int kk = 0; kk < K; kk += 32) {
    v16bf a[2];
#pragma unroll
    for (int mi = 0; mi < 2; ++mi) {
      v8bf lo = *(const v8bf*)(xr[mi] + kk + h * 8);
      v8bf hi = *(const v8bf*)(xr[mi] + kk + 16 + h * 8);
      a[mi] = join8(lo, hi);
    }
    __builtin_prefetch(xr[0] + kk + 64, 0, 1);     // global_prefetch_b8
#pragma unroll
    for (int nt = 0; nt < 4; ++nt) {
      v16bf b = join8(*(const v8bf*)(wr[nt] + kk + h * 8),
                      *(const v8bf*)(wr[nt] + kk + 16 + h * 8));
#pragma unroll
      for (int mi = 0; mi < 2; ++mi)
        acc[mi][nt] = __builtin_amdgcn_wmma_f32_16x16x32_bf16(
            false, a[mi], false, b, (short)0, acc[mi][nt], false, false);
    }
  }

  // C layout: lane n = l16, rows m = j + 8*h
#pragma unroll
  for (int nt = 0; nt < 4; ++nt) {
    const int ngl = nbase + nt * 16 + l16;
    const int hh = ngl >> 6, d = ngl & 63;
#pragma unroll
    for (int mi = 0; mi < 2; ++mi) {
#pragma unroll
      for (int j = 0; j < 8; ++j) {
        int m  = mbase + mi * 16 + j + 8 * h;
        int b_ = m >> 11, t = m & 2047;
        size_t idx = (mode == 0)
            ? (((size_t)b_ * 16 + hh) * 2048 + t) * 64 + d
            : (((size_t)b_ * 16 + hh) * 64 + d) * 2048 + t;
        Y[idx] = (bf16_t)acc[mi][nt][j];
      }
    }
  }
}

// ---------------------------------------------------------------- attention
// Q,K: [BH][T][64] bf16;  Vt: [BH][64][T] bf16 (transposed);  O: [B][T][C] f32
// Block = 128 threads = 4 waves covering 64 Q rows.  Per 64-key block the K/V
// tiles are staged into LDS by the Tensor Data Mover (one 16-row quarter per
// wave), then all waves consume them via WMMA.  Loop count is uniform across
// the workgroup (blocks past a wave's causal limit are fully masked -> no-op).
#define KOFF 0
#define VOFF 8192
#define POFF 16384
#define ATTN_LDS 24576

__global__ void __launch_bounds__(128)
flash_attn(const bf16_t* __restrict__ Q, const bf16_t* __restrict__ Kv,
           const bf16_t* __restrict__ Vt, float* __restrict__ O) {
  extern __shared__ char smem[];
  const int lane  = threadIdx.x & 31;
  const int wv    = threadIdx.x >> 5;
  const int h     = lane >> 4, l16 = lane & 15;
  const int bh    = blockIdx.x;
  const int qbase = blockIdx.y * 64 + wv * 16;
  const int T = 2048, Hs = 64;
  const float scale = 0.125f;  // 64^-0.5

  const bf16_t* q_ptr = Q  + (size_t)bh * T * Hs;
  const bf16_t* k_ptr = Kv + (size_t)bh * T * Hs;
  const bf16_t* v_ptr = Vt + (size_t)bh * Hs * T;

  bf16_t* Ksh = (bf16_t*)(smem + KOFF);   // [64 keys][64 d]
  bf16_t* Vsh = (bf16_t*)(smem + VOFF);   // [64 d][64 keys]
  bf16_t* Psh = (bf16_t*)(smem + POFF);   // [4 waves][16 rows][64 keys]

  // Q A-fragments (K-dim = 64 -> 2 frags), loaded once.
  v16bf aQ[2];
  {
    const bf16_t* qr = q_ptr + (size_t)(qbase + l16) * Hs;
#pragma unroll
    for (int f = 0; f < 2; ++f) {
      v8bf lo = *(const v8bf*)(qr + f * 32 + h * 8);
      v8bf hi = *(const v8bf*)(qr + f * 32 + 16 + h * 8);
      aQ[f] = join8(lo, hi);
    }
  }

  float mrow[8], lrow[8];
#pragma unroll
  for (int j = 0; j < 8; ++j) { mrow[j] = -__builtin_inff(); lrow[j] = 0.f; }
  v8f acc[4] = {};

  const int nkb = blockIdx.y + 1;               // uniform 64-key block count
  const int wvu = __builtin_amdgcn_readfirstlane(wv);

  for (int kb = 0; kb < nkb; ++kb) {
    const int key0 = kb * 64;

    __syncthreads();                            // tiles free to overwrite
#if HAVE_TDM
    // Each wave DMAs a 16-row quarter of the K tile and of the V tile.
    tdm_load_2d(k_ptr + (size_t)(key0 + wvu * 16) * Hs, KOFF + wvu * 2048,
                /*dim0=*/64, /*dim1=*/2048, /*tile0=*/64, /*tile1=*/16,
                /*stride0=*/64);
    tdm_load_2d(v_ptr + (size_t)(wvu * 16) * T + key0, VOFF + wvu * 2048,
                /*dim0=*/2048, /*dim1=*/64, /*tile0=*/64, /*tile1=*/16,
                /*stride0=*/2048);
    __builtin_amdgcn_s_wait_tensorcnt(0);
#else
    {   // cooperative staging fallback: 128 threads x 32B per tile
      int r = threadIdx.x >> 1, c = (threadIdx.x & 1) * 32;
      *(v16bf*)(Ksh + r * 64 + c) =
          *(const v16bf*)(k_ptr + (size_t)(key0 + r) * Hs + c);
      *(v16bf*)(Vsh + r * 64 + c) =
          *(const v16bf*)(v_ptr + (size_t)r * T + key0 + c);
    }
#endif
    __syncthreads();                            // all quarters visible

    // ---- S = Q @ K^T : four 16-key column tiles, K-dim = 64 (2 WMMAs each)
    v8f s[4];
#pragma unroll
    for (int t = 0; t < 4; ++t) {
      const bf16_t* kr = Ksh + (t * 16 + l16) * 64;
      v16bf b0 = join8(*(const v8bf*)(kr + h * 8),
                       *(const v8bf*)(kr + h * 8 + 16));
      v16bf b1 = join8(*(const v8bf*)(kr + 32 + h * 8),
                       *(const v8bf*)(kr + 32 + h * 8 + 16));
      v8f z = {};
      z = __builtin_amdgcn_wmma_f32_16x16x32_bf16(false, aQ[0], false, b0,
                                                  (short)0, z, false, false);
      z = __builtin_amdgcn_wmma_f32_16x16x32_bf16(false, aQ[1], false, b1,
                                                  (short)0, z, false, false);
      s[t] = z;
    }

    // ---- causal mask + scale + row max (row spans the 16-lane group)
    float mnew[8];
#pragma unroll
    for (int j = 0; j < 8; ++j) {
      const int qrow = qbase + j + 8 * h;
      float pm = -__builtin_inff();
#pragma unroll
      for (int t = 0; t < 4; ++t) {
        float sv = s[t][j] * scale;
        if (key0 + t * 16 + l16 > qrow) sv = -__builtin_inff();
        s[t][j] = sv;
        pm = fmaxf(pm, sv);
      }
#pragma unroll
      for (int msk = 1; msk < 16; msk <<= 1)
        pm = fmaxf(pm, __shfl_xor(pm, msk, 32));
      mnew[j] = fmaxf(mrow[j], pm);
    }

    // ---- online softmax update
    float e[4][8];
#pragma unroll
    for (int j = 0; j < 8; ++j) {
      float f  = __expf(mrow[j] - mnew[j]);   // 0 on first block (m = -inf)
      float ps = 0.f;
#pragma unroll
      for (int t = 0; t < 4; ++t) {
        float ev = __expf(s[t][j] - mnew[j]); // masked -> exp(-inf) = 0
        e[t][j] = ev;
        ps += ev;
      }
#pragma unroll
      for (int msk = 1; msk < 16; msk <<= 1)
        ps += __shfl_xor(ps, msk, 32);
      lrow[j] = lrow[j] * f + ps;
      mrow[j] = mnew[j];
#pragma unroll
      for (int k = 0; k < 4; ++k) acc[k][j] *= f;
    }

    // ---- P (C-layout) -> LDS -> A-layout fragments (per-wave region)
#pragma unroll
    for (int t = 0; t < 4; ++t)
#pragma unroll
      for (int j = 0; j < 8; ++j)
        Psh[(wv * 16 + j + 8 * h) * 64 + t * 16 + l16] = (bf16_t)e[t][j];

    asm volatile("s_wait_dscnt 0" ::: "memory");   // same-wave LDS RAW fence

    v16bf aP[2];
#pragma unroll
    for (int f = 0; f < 2; ++f) {
      const bf16_t* pr = Psh + (wv * 16 + l16) * 64 + f * 32;
      v8bf lo = *(const v8bf*)(pr + h * 8);
      v8bf hi = *(const v8bf*)(pr + 16 + h * 8);
      aP[f] = join8(lo, hi);
    }

    // ---- O += P(16x64) @ V-block(64x64)
#pragma unroll
    for (int k = 0; k < 4; ++k) {
#pragma unroll
      for (int f = 0; f < 2; ++f) {
        const bf16_t* vr = Vsh + (k * 16 + l16) * 64 + f * 32 + h * 8;
        v16bf bV = join8(*(const v8bf*)vr, *(const v8bf*)(vr + 16));
        acc[k] = __builtin_amdgcn_wmma_f32_16x16x32_bf16(
            false, aP[f], false, bV, (short)0, acc[k], false, false);
      }
    }
  }

  // ---- epilogue: normalize and scatter to [B][T][H*64] f32
  const int b_ = bh >> 4, hh = bh & 15;
  float rl[8];
#pragma unroll
  for (int j = 0; j < 8; ++j) rl[j] = 1.0f / lrow[j];
#pragma unroll
  for (int k = 0; k < 4; ++k) {
    int d = k * 16 + l16;
#pragma unroll
    for (int j = 0; j < 8; ++j) {
      int t = qbase + j + 8 * h;
      O[((size_t)b_ * 2048 + t) * 1024 + hh * 64 + d] = acc[k][j] * rl[j];
    }
  }
}

// ---------------------------------------------------------------- launcher
extern "C" void kernel_launch(void* const* d_in, const int* in_sizes, int n_in,
                              void* d_out, int out_size, void* d_ws,
                              size_t ws_size, hipStream_t stream) {
  (void)in_sizes; (void)n_in; (void)out_size; (void)ws_size;
  const float* x  = (const float*)d_in[0];
  const float* Wq = (const float*)d_in[1];
  const float* Wk = (const float*)d_in[2];
  const float* Wv = (const float*)d_in[3];
  float* out = (float*)d_out;

  const int B = 4, T = 2048, C = 1024, H = 16;
  const size_t M = (size_t)B * T;  // 8192

  size_t off = 0;
  auto carve = [&](size_t bytes) {
    void* p = (char*)d_ws + off;
    off += (bytes + 255) & ~(size_t)255;
    return p;
  };
  bf16_t* xb  = (bf16_t*)carve(M * C * 2);            // 16 MB
  bf16_t* wqb = (bf16_t*)carve((size_t)C * C * 2);    //  2 MB
  bf16_t* wkb = (bf16_t*)carve((size_t)C * C * 2);
  bf16_t* wvb = (bf16_t*)carve((size_t)C * C * 2);
  bf16_t* qb  = (bf16_t*)carve(M * C * 2);            // [B][H][T][64]
  bf16_t* kbf = (bf16_t*)carve(M * C * 2);            // [B][H][T][64]
  bf16_t* vtb = (bf16_t*)carve(M * C * 2);            // [B][H][64][T]

  const int nx = (int)(M * C);
  const int nw = C * C;
  cvt_f32_bf16<<<(nx + 255) / 256, 256, 0, stream>>>(x,  xb,  nx);
  cvt_f32_bf16<<<(nw + 255) / 256, 256, 0, stream>>>(Wq, wqb, nw);
  cvt_f32_bf16<<<(nw + 255) / 256, 256, 0, stream>>>(Wk, wkb, nw);
  cvt_f32_bf16<<<(nw + 255) / 256, 256, 0, stream>>>(Wv, wvb, nw);

  dim3 pgrid((unsigned)(M / 32), (unsigned)(C / 256));  // 256 x 4
  proj_gemm<<<pgrid, 128, 0, stream>>>(xb, wqb, qb,  0);
  proj_gemm<<<pgrid, 128, 0, stream>>>(xb, wkb, kbf, 0);
  proj_gemm<<<pgrid, 128, 0, stream>>>(xb, wvb, vtb, 1);  // V stored transposed

  dim3 agrid((unsigned)(B * H), (unsigned)(T / 64));      // 64 x 32
  flash_attn<<<agrid, 128, ATTN_LDS, stream>>>(qb, kbf, vtb, out);
}
